// ClusteringLayer_43224550868109
// MI455X (gfx1250) — compile-verified, matrix operand order checked by reading, not run on hardware
//
#include <hip/hip_runtime.h>

// ---------------------------------------------------------------------------
// Deep-embedded-clustering layer, MI455X (gfx1250, wave32, WMMA bf16 path).
//   inputs   [N=65536, D=256] f32
//   clusters [K=256,   D=256] f32
//   out: q [N,K] f32  ++  new_clusters [K,D] f32   (flat concat)
// ---------------------------------------------------------------------------

#define N_SAMPLES 65536
#define DIMS      256
#define KCL       256
#define ROWS_PER_BLOCK 128
#define KSTEP     32
#define NKSTEPS   (DIMS / KSTEP)   // 8
#define CTILES    (KCL / 16)       // 16
#define THRESH    0.1f

typedef __attribute__((ext_vector_type(16))) __bf16 v16bf;
typedef __attribute__((ext_vector_type(8)))  float  v8f;

// f32 -> bf16, round-to-nearest-even, stored as raw u16 so LDS/global staging
// needs no __bf16 arithmetic; fragments are reinterpreted as v16bf at load.
static __device__ __forceinline__ unsigned short f2bf(float f) {
    union { float f; unsigned u; } v; v.f = f;
    unsigned r = v.u + 0x7FFFu + ((v.u >> 16) & 1u);
    return (unsigned short)(r >> 16);
}

// ISA 7.12.2: 16-bit A/B fragment, lanes 0-15 hold K(e)= {0..7,16..23},
// lanes 16-31 hold K(e)= {8..15,24..31}; invert: d' (0..31) -> (laneHi, e).
static __device__ __forceinline__ int frag_lane(int mn, int dp) {
    return (((dp >> 3) & 1) << 4) | (mn & 15);
}
static __device__ __forceinline__ int frag_elem(int dp) {
    return (dp & 7) | (((dp >> 4) & 1) << 3);
}

// ---------------------------------------------------------------------------
// Prep: c2[k] = ||cluster_k||^2, colsum := 0, centroid accum := 0,
// and pack clusters into WMMA-B fragment order (bf16) -> ws.
// Grid: KCL blocks x DIMS threads.
// ---------------------------------------------------------------------------
__global__ void __launch_bounds__(DIMS) prep_kernel(
    const float* __restrict__ clusters,
    float* __restrict__ c2, float* __restrict__ colsum,
    unsigned short* __restrict__ packed, float* __restrict__ cent)
{
    const int c = blockIdx.x;   // cluster
    const int t = threadIdx.x;  // dim
    float v = clusters[c * DIMS + t];

    __shared__ float red[DIMS];
    red[t] = v * v;
    __syncthreads();
    for (int s = DIMS / 2; s > 0; s >>= 1) {
        if (t < s) red[t] += red[t + s];
        __syncthreads();
    }
    if (t == 0) { c2[c] = red[0]; colsum[c] = 0.f; }

    // packed layout: [ks][ct][lane(32)][e(16)] bf16
    const int ks = t >> 5, dp = t & 31;
    const int ct = c >> 4;
    packed[((ks * CTILES + ct) * 32 + frag_lane(c, dp)) * 16 + frag_elem(dp)] = f2bf(v);

    cent[c * DIMS + t] = 0.f;   // zero centroid accumulator (lives in d_out)
}

// ---------------------------------------------------------------------------
// Main fused kernel: GEMM (bf16 WMMA, f32 acc) -> d2 -> q -> row softmax-ish
// normalize -> argmax/threshold -> atomic centroid scatter.
// Grid: N/128 blocks x 256 threads (8 wave32).
// ---------------------------------------------------------------------------
__global__ void __launch_bounds__(256) cluster_main(
    const float* __restrict__ in, const float* __restrict__ c2g,
    const unsigned short* __restrict__ packed,
    float* __restrict__ qout, float* __restrict__ cent,
    float* __restrict__ colsum)
{
    __shared__ __attribute__((aligned(32))) unsigned short As[8 * 32 * 16];       //  8 KB
    __shared__ __attribute__((aligned(32))) unsigned short Bs[CTILES * 32 * 16];  // 16 KB
    __shared__ float x2s[ROWS_PER_BLOCK];
    __shared__ float c2s[KCL];
    __shared__ float wS[ROWS_PER_BLOCK];
    __shared__ int   argS[ROWS_PER_BLOCK];

    const int t      = threadIdx.x;
    const int lane   = t & 31;
    const int wid    = t >> 5;        // wave id = 16-row slab id (0..7)
    const int laneLo = lane & 15;
    const int laneHi = lane >> 4;
    const int rowBase = blockIdx.x * ROWS_PER_BLOCK;

    c2s[t] = c2g[t];  // blockDim == KCL

    // ---- per-row ||x||^2 (2 threads per row, 128 f32 each) ----
    {
        const int lr = t >> 1;
        const int d0 = (t & 1) * 128;
        const float4* p = (const float4*)&in[(size_t)(rowBase + lr) * DIMS + d0];
        float s = 0.f;
        #pragma unroll
        for (int i = 0; i < 32; ++i) {
            float4 x = p[i];
            s += x.x * x.x + x.y * x.y + x.z * x.z + x.w * x.w;
        }
        s += __shfl_xor(s, 1);
        if ((t & 1) == 0) x2s[lr] = s;
    }
    __syncthreads();

    // ---- GEMM: 128 rows x 256 cols, K=256 in 8 steps of 32 (bf16) ----
    v8f acc[CTILES];
    #pragma unroll
    for (int ct = 0; ct < CTILES; ++ct)
        #pragma unroll
        for (int i = 0; i < 8; ++i) acc[ct][i] = 0.f;

    for (int ks = 0; ks < NKSTEPS; ++ks) {
        // stage A tile: f32 -> bf16, fragment order [slab][lane][e]
        {
            const int r   = t >> 1;
            const int dlo = (t & 1) * 16;
            const float* src = &in[(size_t)(rowBase + r) * DIMS + ks * KSTEP + dlo];
            if (ks + 1 < NKSTEPS) __builtin_prefetch(src + KSTEP, 0, 1);
            const int slab = r >> 4;
            #pragma unroll
            for (int j = 0; j < 16; ++j) {
                const int dp = dlo + j;
                As[(slab * 32 + frag_lane(r, dp)) * 16 + frag_elem(dp)] = f2bf(src[j]);
            }
        }
        // stage B tile: straight 16 KB copy of pre-packed fragments
        {
            const float4* src = (const float4*)(packed + (size_t)ks * (CTILES * 32 * 16));
            float4* dst = (float4*)Bs;
            #pragma unroll
            for (int i = 0; i < 4; ++i) dst[t * 4 + i] = src[t * 4 + i];
        }
        __syncthreads();

        const v16bf a = *(const v16bf*)&As[(wid * 32 + lane) * 16];
        #pragma unroll
        for (int ct = 0; ct < CTILES; ++ct) {
            const v16bf b = *(const v16bf*)&Bs[(ct * 32 + lane) * 16];
            acc[ct] = __builtin_amdgcn_wmma_f32_16x16x32_bf16(
                false, a, false, b, (short)0, acc[ct], false, false);
        }
        __syncthreads();
    }

    // ---- epilogue: d2 -> q, row-normalize, argmax + threshold ----
    // C layout: VGPR r holds rows (r | r+8) across lane halves, cols = lanes.
    #pragma unroll
    for (int r = 0; r < 8; ++r) {
        const int lrow = wid * 16 + laneHi * 8 + r;   // local row 0..127
        const float x2v = x2s[lrow];
        float s = 0.f, mx = -3.4e38f; int arg = 0;
        #pragma unroll
        for (int ct = 0; ct < CTILES; ++ct) {
            const float c2v = c2s[ct * 16 + laneLo];
            const float d2  = x2v + c2v - 2.f * acc[ct][r];
            const float qv  = 1.f / (1.f + d2);       // ALPHA = 1
            acc[ct][r] = qv;
            s += qv;
            if (qv > mx) { mx = qv; arg = ct * 16 + laneLo; }
        }
        // reduce across the 16 lanes of this half (xor stays in-half)
        #pragma unroll
        for (int off = 8; off >= 1; off >>= 1) {
            const float os = __shfl_xor(s, off);
            const float om = __shfl_xor(mx, off);
            const int   oa = __shfl_xor(arg, off);
            s += os;
            if (om > mx || (om == mx && oa < arg)) { mx = om; arg = oa; }
        }
        const float inv  = 1.f / s;
        const int   grow = rowBase + lrow;
        #pragma unroll
        for (int ct = 0; ct < CTILES; ++ct)
            qout[(size_t)grow * KCL + ct * 16 + laneLo] = acc[ct][r] * inv;
        if (laneLo == 0) {
            const float qm = mx * inv + 1e-20f;
            wS[lrow]   = (qm > THRESH) ? qm : 0.f;    // relu(h-T)+sign*T == h if h>T else 0
            argS[lrow] = arg;
        }
    }
    __syncthreads();

    // ---- thresholded centroid scatter: cent[arg] += w * x ----
    {
        const int lr = t >> 1;
        const int d0 = (t & 1) * 128;
        const float w = wS[lr];
        if (w > 0.f) {
            const int a = argS[lr];
            const float4* p = (const float4*)&in[(size_t)(rowBase + lr) * DIMS + d0];
            float* cb = cent + (size_t)a * DIMS + d0;
            #pragma unroll
            for (int i = 0; i < 32; ++i) {
                float4 x = p[i];
                atomicAdd(cb + i * 4 + 0, w * x.x);
                atomicAdd(cb + i * 4 + 1, w * x.y);
                atomicAdd(cb + i * 4 + 2, w * x.z);
                atomicAdd(cb + i * 4 + 3, w * x.w);
            }
            if ((t & 1) == 0) atomicAdd(colsum + a, w);
        }
    }
}

// ---------------------------------------------------------------------------
// Finalize: column-normalize centroids (0/0 -> NaN matches reference).
// ---------------------------------------------------------------------------
__global__ void __launch_bounds__(DIMS) finalize_kernel(
    float* __restrict__ cent, const float* __restrict__ colsum)
{
    const int k = blockIdx.x, d = threadIdx.x;
    cent[(size_t)k * DIMS + d] = cent[(size_t)k * DIMS + d] / colsum[k];
}

// ---------------------------------------------------------------------------
extern "C" void kernel_launch(void* const* d_in, const int* in_sizes, int n_in,
                              void* d_out, int out_size, void* d_ws, size_t ws_size,
                              hipStream_t stream) {
    const float* inputs   = (const float*)d_in[0];   // [N, D]
    const float* clusters = (const float*)d_in[1];   // [K, D]

    float* qout = (float*)d_out;                              // [N, K]
    float* cent = qout + (size_t)N_SAMPLES * KCL;             // [K, D]

    // ws: c2 (1 KB) | colsum (1 KB) | packed bf16 clusters (128 KB)
    float* c2     = (float*)d_ws;
    float* colsum = c2 + KCL;
    unsigned short* packed = (unsigned short*)(colsum + KCL);

    prep_kernel<<<KCL, DIMS, 0, stream>>>(clusters, c2, colsum, packed, cent);
    cluster_main<<<N_SAMPLES / ROWS_PER_BLOCK, 256, 0, stream>>>(
        inputs, c2, packed, qout, cent, colsum);
    finalize_kernel<<<KCL, DIMS, 0, stream>>>(cent, colsum);
}
